// AttentionUpdateGRU_77326591197309
// MI455X (gfx1250) — matrix operand order repaired and checked by reading, not compiled, not run on hardware
//
#include <hip/hip_runtime.h>
#include <hip/hip_bf16.h>

// ---------------------------------------------------------------------------
// AttentionUpdateGRU on gfx1250 (MI455X), wave32 + v_wmma_f32_16x16x32_bf16
//   B=1024, T=200, E=256, U=256, 3U=768
// 64 blocks x 512 threads (16 waves). Block owns a 16-row batch tile and
// scans T sequentially; wave w owns gate columns {16w..16w+15} in each of
// the three gate regions, so the nonlinear update is entirely lane-local.
// Weights are pre-packed to bf16 WMMA-B fragment order (L2-resident); an
// opaque per-iteration offset stops LICM from hoisting (and then spilling)
// the weight-fragment loads out of the 200-step loop.
// ---------------------------------------------------------------------------

#define B_ 1024
#define T_ 200
#define E_ 256
#define U_ 256
#define NU_ 768            // 3*U
#define HSTR 264           // padded bf16 row stride for LDS h tile (16B-aligned rows)
#define WPK_ELEMS 196608   // 8 kchunks * 48 ntiles * 32 lanes * 16 elems (per matrix)

typedef __attribute__((ext_vector_type(16))) __bf16 v16bf;
typedef __attribute__((ext_vector_type(8)))  float  v8f;

union BF16x16 {
    v16bf          v;
    unsigned short us[16];
    uint4          q[2];
};

__device__ __forceinline__ unsigned short f2bf(float f) {
    unsigned int b = __builtin_bit_cast(unsigned int, f);
    b += 0x7FFFu + ((b >> 16) & 1u);           // round-to-nearest-even
    return (unsigned short)(b >> 16);
}

__device__ __forceinline__ float hsig(float x) {
    return fminf(fmaxf(0.2f * x + 0.5f, 0.0f), 1.0f);
}

// Load a pre-packed B fragment: 32 contiguous bytes per lane -> 2x b128 loads
__device__ __forceinline__ v16bf ldB(const unsigned short* __restrict__ Wp,
                                     int g, int c, int nt, int lane) {
    BF16x16 b;
    const uint4* p = reinterpret_cast<const uint4*>(
        Wp + (size_t)g * WPK_ELEMS + ((((c * 48) + nt) * 32 + lane) << 4));
    b.q[0] = p[0];
    b.q[1] = p[1];
    return b.v;
}

// ---------------------------------------------------------------------------
// Setup: convert kernel / recurrent_kernel (f32 row-major [256,768]) into
// bf16 WMMA-B fragment order:  Wp[g][c][nt][lane][e]
//   k = c*32 + 16*(e>>3) + 8*(lane>>4) + (e&7) ,  n = nt*16 + (lane&15)
// ---------------------------------------------------------------------------
__global__ void pack_weights_kernel(const float* __restrict__ Wk,
                                    const float* __restrict__ Wr,
                                    unsigned short* __restrict__ Wp) {
    int idx = blockIdx.x * blockDim.x + threadIdx.x;
    if (idx >= 2 * WPK_ELEMS) return;
    int e    = idx & 15;
    int lane = (idx >> 4) & 31;
    int nt   = (idx >> 9) % 48;
    int r2   = (idx >> 9) / 48;
    int c    = r2 & 7;
    int g    = r2 >> 3;
    int k    = c * 32 + ((e >> 3) << 4) + ((lane >> 4) << 3) + (e & 7);
    int n    = nt * 16 + (lane & 15);
    const float* src = g ? Wr : Wk;
    Wp[idx] = f2bf(src[(size_t)k * NU_ + n]);
}

// ---------------------------------------------------------------------------
__global__ __launch_bounds__(512, 1)
void gru_fused_kernel(const float* __restrict__ inputs,
                      const float* __restrict__ alphas,
                      const unsigned char* __restrict__ mask,
                      const float* __restrict__ bias,
                      const unsigned short* __restrict__ Wp,
                      float* __restrict__ out) {
    __shared__ __align__(16) unsigned short Hlds[16 * HSTR];

    const int tid   = threadIdx.x;
    const int lane  = tid & 31;
    const int w     = tid >> 5;            // wave id 0..15
    const int b0    = blockIdx.x * 16;     // batch tile base
    const int mrow  = lane & 15;           // A-operand row for this lane
    const int kb    = (lane >> 4) * 8;     // A-operand K sub-base
    const int nlo   = lane & 15;           // D-operand column within tile
    const int mBase = 8 * (lane >> 4);     // D-operand row base

    // zero the h state tile (h0 = 0)
    for (int i = tid; i < 16 * HSTR; i += 512) Hlds[i] = 0;

    // this wave's n-tiles: z -> nt=w, r -> nt=w+16, hh -> nt=w+32
    const int u = 16 * w + nlo;            // unit column owned by this lane

    // hoisted biases (input_bias + recurrent_bias fold for z/r; separate for h)
    const float bz = bias[u]       + bias[NU_ + u];
    const float br = bias[256 + u] + bias[NU_ + 256 + u];
    const float bx = bias[512 + u];
    const float bq = bias[NU_ + 512 + u];

    float hold[8];
#pragma unroll
    for (int v = 0; v < 8; ++v) hold[v] = 0.0f;

    __syncthreads();

    const float4* Xv = reinterpret_cast<const float4*>(inputs);
    const v8f zero8 = { 0.f, 0.f, 0.f, 0.f, 0.f, 0.f, 0.f, 0.f };
    float* seq_out = out + (size_t)B_ * U_;

    for (int t = 0; t < T_; ++t) {
        // Opaque zero (SGPR): makes weight-fragment addresses look
        // iteration-variant so LICM cannot hoist (and spill) the loads.
        unsigned int tz;
        asm volatile("" : "=s"(tz) : "0"(0u));
        const unsigned short* WpT =
            reinterpret_cast<const unsigned short*>(
                reinterpret_cast<const char*>(Wp) + tz);

        v8f accZ  = zero8;
        v8f accR  = zero8;
        v8f accXH = zero8;
        v8f accRH = zero8;

        const float a_t = alphas[t];
        const size_t xbase = ((size_t)(b0 + mrow) * T_ + t) * E_;

        // ---- GEMM 1: x-projection, A = inputs[b_tile, t, :] (f32 -> bf16) ----
#pragma unroll
        for (int c = 0; c < 8; ++c) {
            BF16x16 A;
            const float4 f0 = Xv[(xbase + c * 32 + kb)          >> 2];
            const float4 f1 = Xv[(xbase + c * 32 + kb + 4)      >> 2];
            const float4 f2 = Xv[(xbase + c * 32 + 16 + kb)     >> 2];
            const float4 f3 = Xv[(xbase + c * 32 + 16 + kb + 4) >> 2];
            A.us[0]  = f2bf(f0.x); A.us[1]  = f2bf(f0.y);
            A.us[2]  = f2bf(f0.z); A.us[3]  = f2bf(f0.w);
            A.us[4]  = f2bf(f1.x); A.us[5]  = f2bf(f1.y);
            A.us[6]  = f2bf(f1.z); A.us[7]  = f2bf(f1.w);
            A.us[8]  = f2bf(f2.x); A.us[9]  = f2bf(f2.y);
            A.us[10] = f2bf(f2.z); A.us[11] = f2bf(f2.w);
            A.us[12] = f2bf(f3.x); A.us[13] = f2bf(f3.y);
            A.us[14] = f2bf(f3.z); A.us[15] = f2bf(f3.w);
            accZ  = __builtin_amdgcn_wmma_f32_16x16x32_bf16(
                false, A.v, false, ldB(WpT, 0, c, w,      lane), (short)0, accZ,  false, false);
            accR  = __builtin_amdgcn_wmma_f32_16x16x32_bf16(
                false, A.v, false, ldB(WpT, 0, c, w + 16, lane), (short)0, accR,  false, false);
            accXH = __builtin_amdgcn_wmma_f32_16x16x32_bf16(
                false, A.v, false, ldB(WpT, 0, c, w + 32, lane), (short)0, accXH, false, false);
        }

        // ---- GEMM 2: recurrent projection, A = h tile (bf16, from LDS) ------
        // z/r accumulate on top of GEMM 1 (x+r fold); h-gate stays separate.
#pragma unroll
        for (int c = 0; c < 8; ++c) {
            BF16x16 A;
            A.q[0] = *reinterpret_cast<const uint4*>(&Hlds[mrow * HSTR + c * 32 + kb]);
            A.q[1] = *reinterpret_cast<const uint4*>(&Hlds[mrow * HSTR + c * 32 + 16 + kb]);
            accZ  = __builtin_amdgcn_wmma_f32_16x16x32_bf16(
                false, A.v, false, ldB(WpT, 1, c, w,      lane), (short)0, accZ,  false, false);
            accR  = __builtin_amdgcn_wmma_f32_16x16x32_bf16(
                false, A.v, false, ldB(WpT, 1, c, w + 16, lane), (short)0, accR,  false, false);
            accRH = __builtin_amdgcn_wmma_f32_16x16x32_bf16(
                false, A.v, false, ldB(WpT, 1, c, w + 32, lane), (short)0, accRH, false, false);
        }

        __syncthreads();   // all waves done reading Hlds for this step

        // ---- lane-local gate math + state update ----------------------------
#pragma unroll
        for (int v = 0; v < 8; ++v) {
            const int m = mBase + v;
            const float zs = accZ[v]  + bz;
            const float rs = accR[v]  + br;
            const float xh = accXH[v] + bx;
            const float rh = accRH[v] + bq;
            const float z  = a_t * hsig(zs);
            const float r  = hsig(rs);
            const float hh = tanhf(xh + r * rh);
            const float hp = hold[v];
            const float hn = hp + z * (hh - hp);
            const unsigned char mk = mask[(size_t)(b0 + m) * T_ + t];
            const float hkeep = mk ? hn : hp;
            hold[v] = hkeep;
            Hlds[m * HSTR + u] = f2bf(hkeep);
            seq_out[((size_t)(b0 + m) * T_ + t) * U_ + u] = hkeep;
        }

        __syncthreads();   // updates visible before next step's A-frag reads
    }

    // last_states = h after final step
#pragma unroll
    for (int v = 0; v < 8; ++v) {
        out[(size_t)(b0 + mBase + v) * U_ + u] = hold[v];
    }
}

// ---------------------------------------------------------------------------
extern "C" void kernel_launch(void* const* d_in, const int* in_sizes, int n_in,
                              void* d_out, int out_size, void* d_ws, size_t ws_size,
                              hipStream_t stream) {
    const float*         inputs  = (const float*)d_in[0];          // [B,T,E]
    const float*         alphas  = (const float*)d_in[1];          // [T]
    const unsigned char* mask    = (const unsigned char*)d_in[2];  // [B,T] bool
    const float*         Wk      = (const float*)d_in[3];          // [E,3U]
    const float*         Wr      = (const float*)d_in[4];          // [U,3U]
    const float*         bias    = (const float*)d_in[5];          // [2,3U]
    float*               out     = (float*)d_out;                  // [B,U] ++ [B,T,U]
    unsigned short*      Wp      = (unsigned short*)d_ws;          // 768 KB packed bf16

    // Phase 0: pack weights into per-lane WMMA-B fragment order (bf16)
    {
        const int total  = 2 * WPK_ELEMS;
        const int blocks = (total + 255) / 256;
        pack_weights_kernel<<<blocks, 256, 0, stream>>>(Wk, Wr, Wp);
    }

    // Phase 1: fused input-projection + recurrent scan
    gru_fused_kernel<<<B_ / 16, 512, 0, stream>>>(inputs, alphas, mask, bias, Wp, out);
}
